// RowSelfAttention_22273700397529
// MI455X (gfx1250) — matrix operand order repaired and checked by reading, not compiled, not run on hardware
//
#include <hip/hip_runtime.h>
#include <hip/hip_bf16.h>

typedef __attribute__((ext_vector_type(16))) __bf16 v16bf;
typedef __attribute__((ext_vector_type(8)))  float  v8f;
typedef unsigned short u16;

struct alignas(16) U16x8 { u16 d[8]; };
union FragU { v16bf v; U16x8 h[2]; };

#define T_TOK 32768
#define EDIM  768
#define HN    24      // heads * n = 12*2
#define SEQ   256
#define RD    4096    // r*d = 64*64
#define KCHUNK 64     // K extent staged in LDS per buffer
#define BLK_N  128    // block tile N
#define USE_ASYNC_COPY 1   // 1: global_load_async_to_lds_b128 path, 0: load+ds_store fallback

__device__ __forceinline__ u16 f32_to_bf16(float f) {
  unsigned int u = __float_as_uint(f);
  u += 0x7FFFu + ((u >> 16) & 1u);   // round-to-nearest-even
  return (u16)(u >> 16);
}

// A fragment: 16x32 bf16. lane L holds row (L&15); elem j -> K = 16*(j>=8) + 8*(L>=16) + (j&7)
__device__ __forceinline__ v16bf load_a_frag(const u16* __restrict__ A, int lda,
                                             int m0, int k0, int lane) {
  const u16* p = A + (long)(m0 + (lane & 15)) * lda + (k0 + ((lane >> 4) << 3));
  FragU u;
  u.h[0] = *(const U16x8*)(p);
  u.h[1] = *(const U16x8*)(p + 16);
  return u.v;
}

// B fragment from LDS chunk bsh[128][KCHUNK]: lane L holds col (L&15); elem j -> K = 16*(j>=8) + 8*(L>=16) + (j&7)
__device__ __forceinline__ v16bf lds_b_frag(const u16* bsh, int n_local, int k_local, int lane) {
  const u16* p = bsh + (n_local + (lane & 15)) * KCHUNK + k_local + ((lane >> 4) << 3);
  FragU u;
  u.h[0] = *(const U16x8*)(p);
  u.h[1] = *(const U16x8*)(p + 16);
  return u.v;
}

// Stage one 128 x KCHUNK bf16 chunk of Bt (pre-transposed [N][K]) into LDS.
// Per thread: row = tid>>1, 64-byte half = tid&1, 4 x 16B transfers.
__device__ __forceinline__ void fill_b_chunk(const u16* __restrict__ Bt, int ldb,
                                             int nblk, int kc0, u16* bsh, int tid) {
  int row  = tid >> 1;
  int half = (tid & 1) * 32;                       // u16 offset (32 u16 = 64B)
  const u16* src = Bt + (long)(nblk + row) * ldb + kc0 + half;
  u16*       dst = bsh + row * KCHUNK + half;
#if USE_ASYNC_COPY
  unsigned           ldso = (unsigned)(size_t)dst;            // low 32 bits = LDS offset
  unsigned long long ga   = (unsigned long long)(size_t)src;
  asm volatile("global_load_async_to_lds_b128 %0, %1, off"            :: "v"(ldso), "v"(ga) : "memory");
  asm volatile("global_load_async_to_lds_b128 %0, %1, off offset:16"  :: "v"(ldso), "v"(ga) : "memory");
  asm volatile("global_load_async_to_lds_b128 %0, %1, off offset:32"  :: "v"(ldso), "v"(ga) : "memory");
  asm volatile("global_load_async_to_lds_b128 %0, %1, off offset:48"  :: "v"(ldso), "v"(ga) : "memory");
#else
#pragma unroll
  for (int q = 0; q < 4; ++q)
    *(U16x8*)(dst + q * 8) = *(const U16x8*)(src + q * 8);
#endif
}

__device__ __forceinline__ void wait_async_copies() {
#if USE_ASYNC_COPY
  asm volatile("s_wait_asynccnt 0" ::: "memory");
#endif
}

// Block: 256 threads = 8 waves (4 row-waves x 2 col-waves). Block tile 128x128.
// Wave tile 32x64 -> 8 accumulators. B double-buffered in LDS via async copies.
__device__ __forceinline__ void gemm_core(const u16* __restrict__ A, int lda,
                                          const u16* __restrict__ Bt, int ldb,
                                          int mblk, int nblk, int Kd,
                                          u16* bsh /*[2][128*KCHUNK]*/, v8f acc[8]) {
  const int tid  = threadIdx.x;
  const int lane = tid & 31;
  const int wave = tid >> 5;
  const int m0   = mblk + (wave & 3) * 32;
  const int n0l  = (wave >> 2) * 64;          // local col base within 128
  v8f z = {};
#pragma unroll
  for (int f = 0; f < 8; ++f) acc[f] = z;

  const int nchunks = Kd / KCHUNK;
  fill_b_chunk(Bt, ldb, nblk, 0, bsh, tid);
  wait_async_copies();
  __syncthreads();

  for (int kc = 0; kc < nchunks; ++kc) {
    u16* cur = bsh + (kc & 1) * (BLK_N * KCHUNK);
    if (kc + 1 < nchunks)
      fill_b_chunk(Bt, ldb, nblk, (kc + 1) * KCHUNK,
                   bsh + ((kc + 1) & 1) * (BLK_N * KCHUNK), tid);
#pragma unroll
    for (int ks = 0; ks < KCHUNK; ks += 32) {
      int k0 = kc * KCHUNK + ks;
      v16bf a0 = load_a_frag(A, lda, m0,      k0, lane);
      v16bf a1 = load_a_frag(A, lda, m0 + 16, k0, lane);
      if (k0 + 32 < Kd)
        __builtin_prefetch(A + (long)(m0 + (lane & 15)) * lda + k0 + 32, 0, 1);
#pragma unroll
      for (int f = 0; f < 4; ++f) {
        v16bf b = lds_b_frag(cur, n0l + 16 * f, ks, lane);
        acc[f]     = __builtin_amdgcn_wmma_f32_16x16x32_bf16(
            false, a0, false, b, (short)0, acc[f],     false, false);
        acc[4 + f] = __builtin_amdgcn_wmma_f32_16x16x32_bf16(
            false, a1, false, b, (short)0, acc[4 + f], false, false);
      }
    }
    wait_async_copies();
    __syncthreads();
  }
}

// acc element (f: col frag, h: row half, v) -> row = m0 + h*16 + 8*(lane>=16) + v,
//                                              col = nblk + n0l + f*16 + (lane&15)

// ---------------- conversion kernels ----------------

__global__ void __launch_bounds__(256) convert_x_kernel(const float* __restrict__ x,
                                                        u16* __restrict__ xb) {
  long i = (long)blockIdx.x * 256 + threadIdx.x;
  xb[i] = f32_to_bf16(x[i]);
}

__global__ void __launch_bounds__(256) convert_w_kernel(
    const float* __restrict__ w0, const float* __restrict__ w1,
    const float* __restrict__ w2, const float* __restrict__ w3,
    u16* __restrict__ o0, u16* __restrict__ o1,
    u16* __restrict__ o2, u16* __restrict__ o3) {
  const float* W; u16* O;
  switch (blockIdx.y) {
    case 0:  W = w0; O = o0; break;
    case 1:  W = w1; O = o1; break;
    case 2:  W = w2; O = o2; break;
    default: W = w3; O = o3; break;
  }
  int idx = blockIdx.x * 256 + threadIdx.x;     // idx = n*768 + k
  int n = idx / EDIM, k = idx % EDIM;
  O[idx] = f32_to_bf16(W[k * EDIM + n]);
}

// ---------------- QKV projection ----------------
__global__ void __launch_bounds__(256) qkv_proj_kernel(
    const u16* __restrict__ X, const u16* __restrict__ Wt,
    const float* __restrict__ bias, u16* __restrict__ out,
    int vlayout, float scale) {
  __shared__ u16 bsh[2 * BLK_N * KCHUNK];
  int lane = threadIdx.x & 31;
  int wave = threadIdx.x >> 5;
  int mblk = blockIdx.x * 128, nblk = blockIdx.y * 128;
  int m0 = mblk + (wave & 3) * 32, n0 = nblk + (wave >> 2) * 64;
  v8f acc[8];
  gemm_core(X, EDIM, Wt, EDIM, mblk, nblk, EDIM, bsh, acc);
#pragma unroll
  for (int f = 0; f < 4; ++f) {
    int col = n0 + f * 16 + (lane & 15);
    float bb = bias[col];
    int hh = col >> 6, dd = col & 63;
#pragma unroll
    for (int h = 0; h < 2; ++h) {
#pragma unroll
      for (int v = 0; v < 8; ++v) {
        int row = m0 + h * 16 + ((lane >> 4) << 3) + v;   // token t = r*512 + i*2 + n
        float val = (acc[h * 4 + f][v] + bb) * scale;
        int rr = row >> 9, ii = (row >> 1) & 255, nn = row & 1;
        long idx;
        if (vlayout)
          idx = ((long)(hh * 2 + nn) * RD + (rr * 64 + dd)) * SEQ + ii;   // Vt
        else
          idx = ((long)(hh * 2 + nn) * SEQ + ii) * RD + (rr * 64 + dd);   // Qs/Ks
        out[idx] = f32_to_bf16(val);
      }
    }
  }
}

// ---------------- attention scores ----------------
__global__ void __launch_bounds__(256) score_kernel(
    const u16* __restrict__ Qs, const u16* __restrict__ Ks, float* __restrict__ S) {
  __shared__ u16 bsh[2 * BLK_N * KCHUNK];
  int lane = threadIdx.x & 31;
  int wave = threadIdx.x >> 5;
  int hn = blockIdx.z;
  const u16* A = Qs + (long)hn * SEQ * RD;
  const u16* B = Ks + (long)hn * SEQ * RD;
  float* out = S + (long)hn * SEQ * SEQ;
  int mblk = blockIdx.x * 128, nblk = blockIdx.y * 128;
  int m0 = mblk + (wave & 3) * 32, n0 = nblk + (wave >> 2) * 64;
  v8f acc[8];
  gemm_core(A, RD, B, RD, mblk, nblk, RD, bsh, acc);
#pragma unroll
  for (int f = 0; f < 4; ++f) {
    int col = n0 + f * 16 + (lane & 15);
#pragma unroll
    for (int h = 0; h < 2; ++h)
#pragma unroll
      for (int v = 0; v < 8; ++v) {
        int row = m0 + h * 16 + ((lane >> 4) << 3) + v;
        out[row * SEQ + col] = acc[h * 4 + f][v];
      }
  }
}

// ---------------- softmax over last dim (256) ----------------
__global__ void __launch_bounds__(256) softmax_kernel(const float* __restrict__ S,
                                                      u16* __restrict__ P) {
  __shared__ float sm[256];
  long row = blockIdx.x;
  int t = threadIdx.x;
  float v = S[row * SEQ + t];
  sm[t] = v; __syncthreads();
  for (int s = 128; s > 0; s >>= 1) { if (t < s) sm[t] = fmaxf(sm[t], sm[t + s]); __syncthreads(); }
  float mx = sm[0]; __syncthreads();
  float e = __expf(v - mx);
  sm[t] = e; __syncthreads();
  for (int s = 128; s > 0; s >>= 1) { if (t < s) sm[t] += sm[t + s]; __syncthreads(); }
  float inv = 1.0f / sm[0];
  P[row * SEQ + t] = f32_to_bf16(e * inv);
}

// ---------------- context ----------------
__global__ void __launch_bounds__(256) ctx_kernel(
    const u16* __restrict__ P, const u16* __restrict__ Vt, u16* __restrict__ Ctx) {
  __shared__ u16 bsh[2 * BLK_N * KCHUNK];
  int lane = threadIdx.x & 31;
  int wave = threadIdx.x >> 5;
  int hn = blockIdx.z;
  int hh = hn >> 1, nn = hn & 1;
  const u16* A = P + (long)hn * SEQ * SEQ;      // [256 x 256]
  const u16* B = Vt + (long)hn * RD * SEQ;      // pre-transposed [4096 x 256]
  int mblk = blockIdx.x * 128, nblk = blockIdx.y * 128;   // i, rd
  int m0 = mblk + (wave & 3) * 32, n0 = nblk + (wave >> 2) * 64;
  v8f acc[8];
  gemm_core(A, SEQ, B, SEQ, mblk, nblk, SEQ, bsh, acc);
#pragma unroll
  for (int f = 0; f < 4; ++f) {
    int col = n0 + f * 16 + (lane & 15);        // rd = r*64+d
    int rr = col >> 6, dd = col & 63;
#pragma unroll
    for (int h = 0; h < 2; ++h)
#pragma unroll
      for (int v = 0; v < 8; ++v) {
        int ii = m0 + h * 16 + ((lane >> 4) << 3) + v;
        long tok = (long)rr * 512 + ii * 2 + nn;
        Ctx[tok * EDIM + hh * 64 + dd] = f32_to_bf16(acc[h * 4 + f][v]);
      }
  }
}

// ---------------- output projection ----------------
__global__ void __launch_bounds__(256) out_proj_kernel(
    const u16* __restrict__ Ctx, const u16* __restrict__ WoT,
    const float* __restrict__ bias, float* __restrict__ out) {
  __shared__ u16 bsh[2 * BLK_N * KCHUNK];
  int lane = threadIdx.x & 31;
  int wave = threadIdx.x >> 5;
  int mblk = blockIdx.x * 128, nblk = blockIdx.y * 128;
  int m0 = mblk + (wave & 3) * 32, n0 = nblk + (wave >> 2) * 64;
  v8f acc[8];
  gemm_core(Ctx, EDIM, WoT, EDIM, mblk, nblk, EDIM, bsh, acc);
#pragma unroll
  for (int f = 0; f < 4; ++f) {
    int col = n0 + f * 16 + (lane & 15);
    float bb = bias[col];
#pragma unroll
    for (int h = 0; h < 2; ++h)
#pragma unroll
      for (int v = 0; v < 8; ++v) {
        int row = m0 + h * 16 + ((lane >> 4) << 3) + v;
        out[(long)row * EDIM + col] = acc[h * 4 + f][v] + bb;
      }
  }
}

extern "C" void kernel_launch(void* const* d_in, const int* in_sizes, int n_in,
                              void* d_out, int out_size, void* d_ws, size_t ws_size,
                              hipStream_t stream) {
  const float* x  = (const float*)d_in[0];
  const float* Wq = (const float*)d_in[1];
  const float* bq = (const float*)d_in[2];
  const float* Wk = (const float*)d_in[3];
  const float* bk = (const float*)d_in[4];
  const float* Wv = (const float*)d_in[5];
  const float* bv = (const float*)d_in[6];
  const float* Wo = (const float*)d_in[7];
  const float* bo = (const float*)d_in[8];
  float* out = (float*)d_out;

  char* ws = (char*)d_ws;
  auto alloc = [&](size_t bytes) {
    char* p = ws;
    ws += (bytes + 255) & ~(size_t)255;
    return p;
  };
  const size_t TE = (size_t)T_TOK * EDIM;
  u16*   xb  = (u16*)  alloc(TE * 2);                       // x bf16 [T x 768]
  u16*   WqT = (u16*)  alloc((size_t)EDIM * EDIM * 2);
  u16*   WkT = (u16*)  alloc((size_t)EDIM * EDIM * 2);
  u16*   WvT = (u16*)  alloc((size_t)EDIM * EDIM * 2);
  u16*   WoT = (u16*)  alloc((size_t)EDIM * EDIM * 2);
  u16*   Qs  = (u16*)  alloc((size_t)HN * SEQ * RD * 2);    // [hn][i][rd]
  u16*   Ks  = (u16*)  alloc((size_t)HN * SEQ * RD * 2);    // [hn][j][rd]
  u16*   Vt  = (u16*)  alloc((size_t)HN * RD * SEQ * 2);    // [hn][rd][j]
  float* Sc  = (float*)alloc((size_t)HN * SEQ * SEQ * 4);   // scores f32
  u16*   Pr  = (u16*)  alloc((size_t)HN * SEQ * SEQ * 2);   // probs bf16
  u16*   Ctx = (u16*)  alloc(TE * 2);                       // context bf16 [T x 768]

  const float scaling = 0.015625f;   // d^-0.5 / sqrt(r) = 0.125 / 8

  convert_x_kernel<<<dim3((unsigned)(TE / 256)), 256, 0, stream>>>(x, xb);
  convert_w_kernel<<<dim3(EDIM * EDIM / 256, 4), 256, 0, stream>>>(
      Wq, Wk, Wv, Wo, WqT, WkT, WvT, WoT);

  dim3 gProj(T_TOK / 128, EDIM / 128);
  qkv_proj_kernel<<<gProj, 256, 0, stream>>>(xb, WqT, bq, Qs, 0, scaling);
  qkv_proj_kernel<<<gProj, 256, 0, stream>>>(xb, WkT, bk, Ks, 0, 1.0f);
  qkv_proj_kernel<<<gProj, 256, 0, stream>>>(xb, WvT, bv, Vt, 1, 1.0f);

  score_kernel<<<dim3(SEQ / 128, SEQ / 128, HN), 256, 0, stream>>>(Qs, Ks, Sc);
  softmax_kernel<<<dim3(HN * SEQ), 256, 0, stream>>>(Sc, Pr);
  ctx_kernel<<<dim3(SEQ / 128, RD / 128, HN), 256, 0, stream>>>(Pr, Vt, Ctx);
  out_proj_kernel<<<gProj, 256, 0, stream>>>(Ctx, WoT, bo, out);
}
